// MambaBlock_67405216743458
// MI455X (gfx1250) — matrix operand order repaired and checked by reading, not compiled, not run on hardware
//
#include <hip/hip_runtime.h>
#include <hip/hip_bf16.h>
#include <math.h>

#ifndef __has_builtin
#define __has_builtin(x) 0
#endif

// Problem constants (match the reference)
#define BB    2
#define TT    1024
#define DM    1024
#define DI    2048           // d_inner
#define DS    64             // d_state
#define DTR   64             // dt_rank
#define MROWS (BB * TT)      // 2048 flattened (b,t) rows
#define XZN   (2 * DI)       // 4096
#define XDN   (DTR + 2 * DS) // 192

typedef __attribute__((ext_vector_type(16))) _Float16 v16h;
typedef __attribute__((ext_vector_type(8)))  _Float16 v8h;
typedef __attribute__((ext_vector_type(8)))  float    v8f;
typedef int i32x4v __attribute__((vector_size(16)));   // b128 payload type

union V16HU { v16h v; v8h h[2]; };

// ---------------------------------------------------------------------------
// Async global->LDS 16B copy (CDNA5 GLOBAL_LOAD_ASYNC_TO_LDS_B128, ASYNCcnt),
// with a synchronous fallback if this toolchain lacks the builtin.
// ---------------------------------------------------------------------------
#if __has_builtin(__builtin_amdgcn_global_load_async_to_lds_b128)
#define HAVE_ASYNC_LDS 1
#else
#define HAVE_ASYNC_LDS 0
#endif

__device__ __forceinline__ void async_copy16(const _Float16* g, _Float16* l) {
#if HAVE_ASYNC_LDS
  __builtin_amdgcn_global_load_async_to_lds_b128(
      (__attribute__((address_space(1))) i32x4v*)g,
      (__attribute__((address_space(3))) i32x4v*)l, 0, 0);
#else
  *(v8h*)l = *(const v8h*)g;                       // global b128 load + ds store
#endif
}

__device__ __forceinline__ void wait_stage() {
#if HAVE_ASYNC_LDS
#if __has_builtin(__builtin_amdgcn_s_wait_asynccnt)
  __builtin_amdgcn_s_wait_asynccnt(0);             // s_wait_asynccnt 0
#endif
#endif
  __syncthreads();
}

// ---------------------------------------------------------------------------
// fp32 -> fp16 conversion (GEMM operand staging)
// ---------------------------------------------------------------------------
__global__ void cvt_f32_to_f16_kernel(const float* __restrict__ src,
                                      _Float16* __restrict__ dst, int n) {
  int i = blockIdx.x * blockDim.x + threadIdx.x;
  if (i < n) dst[i] = (_Float16)src[i];
}

// ---------------------------------------------------------------------------
// LDS double-buffered WMMA GEMM: C[M,N] = A[M,K] * W[N,K]^T (f16 in, f32 acc).
// Block = 256 threads = 8 waves arranged WM x WN (WM*WN == 8).
// Macro tile = (WM*16) x (WN*64); each wave computes 16x64 (4 accumulators).
// K-step = 32. Operand tiles staged global->LDS with async b128 copies
// (straight-line, statically unrolled -> no divergent staging loop),
// double buffered so the copy of step k+1 overlaps the WMMAs of step k.
// mode 0: plain   mode 1: softplus(v + aux[n])   mode 2: v * aux[m] (mask)
// ---------------------------------------------------------------------------
template <int WM, int WN>
__global__ void __launch_bounds__(256)
gemm_wmma_lds(const _Float16* __restrict__ A, const _Float16* __restrict__ W,
              float* __restrict__ C, int M, int N, int K, int mode,
              const float* __restrict__ aux) {
  __shared__ alignas(64) _Float16 As[2][WM * 16][32];
  __shared__ alignas(64) _Float16 Bs[2][WN * 64][32];

  const int tid  = threadIdx.x;
  const int lane = tid & 31;
  const int wid  = tid >> 5;
  const int wm   = wid / WN;
  const int wn   = wid % WN;

  const int mbase = blockIdx.y * (WM * 16);     // grids sized exactly: no tails
  const int nbase = blockIdx.x * (WN * 64);

  constexpr int ACH = WM * 64;    // 16B chunks per A stage (4 per 32-half row)
  constexpr int BCH = WN * 256;   // 16B chunks per B stage

  auto stage = [&](int buf, int kb) {
    // A tile: static trip count per thread
    if (ACH >= 256) {
#pragma unroll
      for (int i = 0; i < ACH / 256; ++i) {
        const int c = tid + i * 256;
        const int row = c >> 2, kc = (c & 3) << 3;
        async_copy16(A + (size_t)(mbase + row) * K + kb + kc, &As[buf][row][kc]);
      }
    } else if (tid < ACH) {
      const int row = tid >> 2, kc = (tid & 3) << 3;
      async_copy16(A + (size_t)(mbase + row) * K + kb + kc, &As[buf][row][kc]);
    }
    // B tile: exact multiple of 256 chunks for both configs
#pragma unroll
    for (int i = 0; i < BCH / 256; ++i) {
      const int c = tid + i * 256;
      const int row = c >> 2, kc = (c & 3) << 3;
      async_copy16(W + (size_t)(nbase + row) * K + kb + kc, &Bs[buf][row][kc]);
    }
  };

  v8f acc[4] = {};

  // Fragment addressing used here (wave32, contiguous k, same order for A and B):
  //  A 16x32 f16: row = lane&15; 16 contiguous K starting at koff,
  //               koff = (lane>=16)*16
  //  B 32x16 f16: col = lane&15; 16 contiguous K starting at (lane>=16)*16
  const int arow  = wm * 16 + (lane & 15);
  const int akoff = (lane >> 4) * 16;
  const int bkoff = (lane >> 4) * 16;
  const int bcol  = lane & 15;

  stage(0, 0);
  wait_stage();

  int buf = 0;
  for (int kb = 0; kb < K; kb += 32) {
    if (kb + 32 < K) stage(buf ^ 1, kb + 32);   // overlaps WMMAs below

    // hoist all fragment loads so the ds_load clause completes once,
    // then the 4 WMMAs issue back-to-back
    V16HU au;
    au.h[0] = *(const v8h*)(&As[buf][arow][akoff]);
    au.h[1] = *(const v8h*)(&As[buf][arow][akoff + 8]);
    const v16h af = au.v;
    v16h bf[4];
#pragma unroll
    for (int s = 0; s < 4; ++s)
      bf[s] = *(const v16h*)(&Bs[buf][wn * 64 + s * 16 + bcol][bkoff]);
#pragma unroll
    for (int s = 0; s < 4; ++s)
      acc[s] = __builtin_amdgcn_wmma_f32_16x16x32_f16(
          /*neg_a=*/false, af, /*neg_b=*/false, bf[s],
          /*c_mod=*/(short)0, acc[s], /*reuse_a=*/false, /*reuse_b=*/false);

    wait_stage();                               // next buffer ready for all waves
    buf ^= 1;
  }

  // D layout: VGPR r -> row r + (lane>=16)*8, col = lane&15
  const int rbase = mbase + wm * 16 + (lane >> 4) * 8;
#pragma unroll
  for (int s = 0; s < 4; ++s) {
    const int gn = nbase + wn * 64 + s * 16 + (lane & 15);
#pragma unroll
    for (int r = 0; r < 8; ++r) {
      const int gm = rbase + r;
      float v = acc[s][r];
      if (mode == 1) {                          // dt: bias + softplus
        v += aux[gn];
        v = (v > 20.f) ? v : log1pf(__expf(v));
      } else if (mode == 2) {                   // out: padding mask per row
        v *= aux[gm];
      }
      C[(size_t)gm * N + gn] = v;
    }
  }
}

// ---------------------------------------------------------------------------
// Causal depthwise conv1d (width 4) + bias + SiLU over the xs half of xz.
// ---------------------------------------------------------------------------
__global__ void conv_silu_kernel(const float* __restrict__ xz,
                                 const float* __restrict__ cw,
                                 const float* __restrict__ cb,
                                 float* __restrict__ xs,
                                 _Float16* __restrict__ xsh) {
  int idx = blockIdx.x * blockDim.x + threadIdx.x;
  if (idx >= MROWS * DI) return;
  int d = idx % DI;
  int m = idx / DI;
  int t = m % TT;
  float acc = cb[d];
#pragma unroll
  for (int j = 0; j < 4; ++j) {
    int tt = t - 3 + j;
    if (tt >= 0) acc += cw[d * 4 + j] * xz[(size_t)(m - 3 + j) * XZN + d];
  }
  float s = acc / (1.f + __expf(-acc));   // SiLU
  xs[idx]  = s;
  xsh[idx] = (_Float16)s;
}

// dt_in slice of x_dbl -> f16 for the dt_proj GEMM
__global__ void cvt_dtin_kernel(const float* __restrict__ xdbl,
                                _Float16* __restrict__ dst) {
  int i = blockIdx.x * blockDim.x + threadIdx.x;
  if (i >= MROWS * DTR) return;
  int m = i / DTR, r = i % DTR;
  dst[i] = (_Float16)xdbl[(size_t)m * XDN + r];
}

// ---------------------------------------------------------------------------
// Selective scan. Grid: (DI/64, B) blocks of 256 threads.
// Lane quad owns one channel; each lane keeps 16 h-states and 16 A entries
// in VGPRs. B_t/C_t double-buffered through LDS -> one barrier per timestep.
// ---------------------------------------------------------------------------
__global__ void __launch_bounds__(256)
ssm_scan_kernel(const float* __restrict__ dt,
                const float* __restrict__ xs,
                const float* __restrict__ xdbl,
                const float* __restrict__ A_log,
                float* __restrict__ ysc) {
  __shared__ float sB[2][DS];
  __shared__ float sC[2][DS];
  const int tid   = threadIdx.x;
  const int chunk = blockIdx.x;           // 0..31
  const int b     = blockIdx.y;           // 0..1
  const int ch    = chunk * 64 + (tid >> 2);
  const int part  = tid & 3;
  const int nb    = part * 16;

  float a[16], h[16];
#pragma unroll
  for (int i = 0; i < 16; ++i) {
    a[i] = -__expf(A_log[(size_t)ch * DS + nb + i]);
    h[i] = 0.f;
  }

  for (int t = 0; t < TT; ++t) {
    const int m   = b * TT + t;
    const int bt  = t & 1;
    if (tid < DS)          sB[bt][tid]      = xdbl[(size_t)m * XDN + DTR + tid];
    else if (tid < 2 * DS) sC[bt][tid - DS] = xdbl[(size_t)m * XDN + DTR + DS + (tid - DS)];
    __syncthreads();

    const float dtv = dt[(size_t)m * DI + ch];
    const float dtx = dtv * xs[(size_t)m * DI + ch];
    float y = 0.f;
#pragma unroll
    for (int i = 0; i < 16; ++i) {
      h[i] = __expf(dtv * a[i]) * h[i] + dtx * sB[bt][nb + i];
      y += h[i] * sC[bt][nb + i];
    }
    y += __shfl_xor(y, 1);
    y += __shfl_xor(y, 2);
    if (part == 0) ysc[(size_t)m * DI + ch] = y;
    // no trailing barrier: next step writes the other LDS buffer
  }
}

// y = (y_scan + D*xs) * silu(z), emitted as f16 for the out_proj GEMM
__global__ void gate_kernel(const float* __restrict__ ysc,
                            const float* __restrict__ xs,
                            const float* __restrict__ xz,
                            const float* __restrict__ Dv,
                            _Float16* __restrict__ yh) {
  int idx = blockIdx.x * blockDim.x + threadIdx.x;
  if (idx >= MROWS * DI) return;
  int d = idx % DI;
  int m = idx / DI;
  float y = ysc[idx] + Dv[d] * xs[idx];
  float z = xz[(size_t)m * XZN + DI + d];
  float g = z / (1.f + __expf(-z));
  yh[idx] = (_Float16)(y * g);
}

// ---------------------------------------------------------------------------
extern "C" void kernel_launch(void* const* d_in, const int* in_sizes, int n_in,
                              void* d_out, int out_size, void* d_ws, size_t ws_size,
                              hipStream_t stream) {
  (void)in_sizes; (void)n_in; (void)out_size; (void)ws_size;
  const float* x        = (const float*)d_in[0];
  const float* mask     = (const float*)d_in[1];
  const float* in_w     = (const float*)d_in[2];
  const float* conv_w   = (const float*)d_in[3];
  const float* conv_b   = (const float*)d_in[4];
  const float* xproj_w  = (const float*)d_in[5];
  const float* dtproj_w = (const float*)d_in[6];
  const float* dtproj_b = (const float*)d_in[7];
  const float* A_log    = (const float*)d_in[8];
  const float* Dv       = (const float*)d_in[9];
  const float* out_w    = (const float*)d_in[10];
  float* out = (float*)d_out;

  char* base = (char*)d_ws;
  size_t off = 0;
  auto alloc = [&](size_t bytes) -> char* {
    char* p = base + off;
    off = (off + bytes + 255) & ~(size_t)255;
    return p;
  };
  _Float16* Xh    = (_Float16*)alloc((size_t)MROWS * DM * 2);
  _Float16* Wih   = (_Float16*)alloc((size_t)XZN * DM * 2);
  _Float16* Wxh   = (_Float16*)alloc((size_t)XDN * DI * 2);
  _Float16* Wdth  = (_Float16*)alloc((size_t)DI * DTR * 2);
  _Float16* Woh   = (_Float16*)alloc((size_t)DM * DI * 2);
  float*    xz    = (float*)   alloc((size_t)MROWS * XZN * 4);
  float*    xs    = (float*)   alloc((size_t)MROWS * DI * 4);
  _Float16* xsh   = (_Float16*)alloc((size_t)MROWS * DI * 2);
  float*    xdbl  = (float*)   alloc((size_t)MROWS * XDN * 4);
  _Float16* dtinh = (_Float16*)alloc((size_t)MROWS * DTR * 2);
  float*    dtf   = (float*)   alloc((size_t)MROWS * DI * 4);
  float*    ysc   = (float*)   alloc((size_t)MROWS * DI * 4);
  _Float16* yh    = (_Float16*)alloc((size_t)MROWS * DI * 2);

  auto cvt = [&](const float* s, _Float16* d, int n) {
    cvt_f32_to_f16_kernel<<<(n + 255) / 256, 256, 0, stream>>>(s, d, n);
  };
  cvt(x,        Xh,   MROWS * DM);
  cvt(in_w,     Wih,  XZN * DM);
  cvt(xproj_w,  Wxh,  XDN * DI);
  cvt(dtproj_w, Wdth, DI * DTR);
  cvt(out_w,    Woh,  DM * DI);

  // macro tile 32x256 (2x4 waves) for wide GEMMs
  auto gemm24 = [&](const _Float16* A, const _Float16* W, float* C,
                    int M, int N, int K, int mode, const float* aux) {
    gemm_wmma_lds<2, 4><<<dim3(N / 256, M / 32), 256, 0, stream>>>(A, W, C, M, N, K, mode, aux);
  };
  // macro tile 128x64 (8x1 waves) for the narrow N=192 GEMM
  auto gemm81 = [&](const _Float16* A, const _Float16* W, float* C,
                    int M, int N, int K, int mode, const float* aux) {
    gemm_wmma_lds<8, 1><<<dim3(N / 64, M / 128), 256, 0, stream>>>(A, W, C, M, N, K, mode, aux);
  };

  // 1) xz = x @ in_proj_w^T                [2048,4096]
  gemm24(Xh, Wih, xz, MROWS, XZN, DM, 0, nullptr);
  // 2) causal depthwise conv + SiLU on xs half
  conv_silu_kernel<<<(MROWS * DI + 255) / 256, 256, 0, stream>>>(xz, conv_w, conv_b, xs, xsh);
  // 3) x_dbl = xs @ x_proj_w^T             [2048,192]
  gemm81(xsh, Wxh, xdbl, MROWS, XDN, DI, 0, nullptr);
  // 4) dt = softplus(dt_in @ dt_proj_w^T + b)   [2048,2048]
  cvt_dtin_kernel<<<(MROWS * DTR + 255) / 256, 256, 0, stream>>>(xdbl, dtinh);
  gemm24(dtinh, Wdth, dtf, MROWS, DI, DTR, 1, dtproj_b);
  // 5) selective scan (sequential over T)
  ssm_scan_kernel<<<dim3(DI / 64, BB), 256, 0, stream>>>(dtf, xs, xdbl, A_log, ysc);
  // 6) gate: (y + D*xs) * silu(z)
  gate_kernel<<<(MROWS * DI + 255) / 256, 256, 0, stream>>>(ysc, xs, xz, Dv, yh);
  // 7) out = y @ out_proj_w^T, masked      [2048,1024]
  gemm24(yh, Woh, out, MROWS, DM, DI, 2, mask);
}
